// DyDCNv2_9156870275199
// MI455X (gfx1250) — matrix operand order repaired and checked by reading, not compiled, not run on hardware
//
#include <hip/hip_runtime.h>
#include <hip/hip_bf16.h>

#define Bz    4
#define CIN   256
#define COUT  256
#define Hs    96
#define Ws    96
#define HW    (Hs * Ws)
#define GROUPS 16
#define EPS   1e-5f

// workspace: [0,128) floats = GN accumulators; packed bf16 weight fragments after.
#define WS_ACC_FLOATS 128
#define PW_DWORDS     (9 * 8 * 16 * 256)   // t * cinChunk * coutTile * (32 lanes * 8 dw)

#define APAD 40                            // Alds row stride (ushorts): 80 B, 16 B aligned

typedef __bf16 v16bf __attribute__((ext_vector_type(16)));
typedef float  v8f   __attribute__((ext_vector_type(8)));

union FragBF16 { v16bf v; unsigned int u[8]; uint4 q[2]; };

__device__ __forceinline__ unsigned short f2bf(float f) {
    unsigned int u = __float_as_uint(f);
    unsigned int r = u + 0x7FFFu + ((u >> 16) & 1u);   // round-to-nearest-even
    return (unsigned short)(r >> 16);
}

// ---------------------------------------------------------------------------
// Kernel 0: zero GN accumulators
// ---------------------------------------------------------------------------
__global__ void zero_ws_kernel(float* ws) {
    int i = threadIdx.x;
    if (i < WS_ACC_FLOATS) ws[i] = 0.0f;
}

// ---------------------------------------------------------------------------
// Kernel 1: pack conv weights into per-lane WMMA B-fragment layout (bf16).
// pw dword index = (((t*8 + c8)*16 + nt)*32 + lane)*8 + v
// ---------------------------------------------------------------------------
__global__ void __launch_bounds__(256)
pack_weight_kernel(const float* __restrict__ weight, unsigned int* __restrict__ pw) {
    const int idx = blockIdx.x * 256 + threadIdx.x;     // 0 .. PW_DWORDS-1
    const int v    = idx & 7;
    const int lane = (idx >> 3) & 31;
    const int nt   = (idx >> 8) & 15;
    const int c8   = (idx >> 12) & 7;
    const int t    = idx >> 15;
    const int n    = nt * 16 + (lane & 15);
    const int hi   = lane >> 4;
    const int k0   = 2 * v + hi * 16;
    const int cin0 = c8 * 32 + k0;
    const size_t base = ((size_t)n * CIN + cin0) * 9 + t;
    const float w0 = weight[base];
    const float w1 = weight[base + 9];                  // cin0 + 1
    pw[idx] = (unsigned int)f2bf(w0) | ((unsigned int)f2bf(w1) << 16);
}

// ---------------------------------------------------------------------------
// Kernel 2: fused modulated-deformable-conv as implicit GEMM on WMMA bf16.
// Block: 256 threads (8 waves). M-tile = 32 pixels (2 x 16-row A sub-tiles),
// N = all 256 couts (2 x 16-col tiles per wave) -> 4 WMMAs per K-step per wave.
// K = 9 taps * 256 cin, stepped by 32.
// ---------------------------------------------------------------------------
__global__ void __launch_bounds__(256)
dcn_wmma_kernel(const float* __restrict__ x,
                const float* __restrict__ offset,
                const float* __restrict__ mask,
                const unsigned int* __restrict__ pw,
                float* __restrict__ out,
                float* __restrict__ ws) {
    __shared__ unsigned short Alds[32][APAD];  // 32 m x 32 k (bf16), padded rows
    __shared__ int   sIdx[4][32];              // 4 bilinear neighbors per pixel
    __shared__ float sWgt[4][32];              // bilinear weight * valid * mask

    const int tid  = threadIdx.x;
    const int lane = tid & 31;
    const int wave = tid >> 5;
    const int hi   = lane >> 4;
    const int nl16 = lane & 15;

    // decode spatial tile: gridDim.x = B * Hs * (Ws/32)
    const int tile  = blockIdx.x;
    const int xTile = tile % (Ws / 32);
    const int yRow  = (tile / (Ws / 32)) % Hs;
    const int b     = tile / ((Ws / 32) * Hs);
    const int xBase = xTile * 32;

    const int nt0 = wave * 2;                  // this wave's two cout tiles
    const int nt1 = wave * 2 + 1;

    v8f acc00 = {}, acc01 = {};                // [mTile][nTile]
    v8f acc10 = {}, acc11 = {};

    for (int t = 0; t < 9; ++t) {
        __syncthreads();
        // --- per-tap sampling coordinates (channel-invariant), 32 pixels ---
        if (tid < 32) {
            const int m  = tid;
            const int xp = xBase + m;
            const int ki = t / 3, kj = t % 3;
            const float offy = offset[((size_t)(b * 18 + 2 * t) * Hs + yRow) * Ws + xp];
            const float offx = offset[((size_t)(b * 18 + 2 * t + 1) * Hs + yRow) * Ws + xp];
            const float mk   = mask[((size_t)(b * 9 + t) * Hs + yRow) * Ws + xp];
            const float py = (float)yRow + (float)(ki - 1) + offy;
            const float px = (float)xp   + (float)(kj - 1) + offx;
            const float y0 = floorf(py);
            const float x0 = floorf(px);
#pragma unroll
            for (int j = 0; j < 4; ++j) {
                const float yy = y0 + (float)(j >> 1);
                const float xx = x0 + (float)(j & 1);
                float w = (1.0f - fabsf(py - yy)) * (1.0f - fabsf(px - xx));
                const bool valid = (yy >= 0.0f) && (yy < (float)Hs) &&
                                   (xx >= 0.0f) && (xx < (float)Ws);
                int yi = min(max((int)yy, 0), Hs - 1);
                int xi = min(max((int)xx, 0), Ws - 1);
                sIdx[j][m] = yi * Ws + xi;
                sWgt[j][m] = valid ? (w * mk) : 0.0f;
            }
        }

        for (int c8 = 0; c8 < 8; ++c8) {
            __syncthreads();   // coords visible / previous fragment reads done

            // --- A tile: 1024 samples, 4 per thread.
            // Lanes within a wave share the channel set and vary the pixel m,
            // so neighbor gathers hit a narrow window of one H*W plane. ---
            {
                const int m  = tid & 31;
                const int c0 = (tid >> 5) * 4;             // channel within chunk
                const int i0 = sIdx[0][m], i1 = sIdx[1][m], i2 = sIdx[2][m], i3 = sIdx[3][m];
                const float w0 = sWgt[0][m], w1 = sWgt[1][m], w2 = sWgt[2][m], w3 = sWgt[3][m];
                const float* xb = x + (size_t)(b * CIN + c8 * 32 + c0) * HW;
                unsigned int pk[2];
#pragma unroll
                for (int j = 0; j < 2; ++j) {
                    const float va = w0 * xb[i0] + w1 * xb[i1] + w2 * xb[i2] + w3 * xb[i3];
                    xb += HW;
                    const float vb = w0 * xb[i0] + w1 * xb[i1] + w2 * xb[i2] + w3 * xb[i3];
                    xb += HW;
                    pk[j] = (unsigned int)f2bf(va) | ((unsigned int)f2bf(vb) << 16);
                }
                *(uint2*)&Alds[m][c0] = make_uint2(pk[0], pk[1]);
            }
            __syncthreads();   // tile visible

            // --- A fragments from LDS (contiguous k = 16*hi + s, matches packed B), 2 M sub-tiles ---
            FragBF16 fa0, fa1;
#pragma unroll
            for (int v = 0; v < 8; ++v) {
                const int ka = 2 * v + hi * 16;
                fa0.u[v] = *(const unsigned int*)&Alds[nl16][ka];
                fa1.u[v] = *(const unsigned int*)&Alds[16 + nl16][ka];
            }
            // --- B fragments straight from pre-packed global (L2-resident) ---
            const int tc = (t * 8 + c8) * 16;
            FragBF16 fb0, fb1;
            {
                const unsigned int* p0 = pw + ((size_t)(tc + nt0) * 32 + lane) * 8;
                const unsigned int* p1 = pw + ((size_t)(tc + nt1) * 32 + lane) * 8;
                fb0.q[0] = *(const uint4*)(p0);
                fb0.q[1] = *(const uint4*)(p0 + 4);
                fb1.q[0] = *(const uint4*)(p1);
                fb1.q[1] = *(const uint4*)(p1 + 4);
            }
            acc00 = __builtin_amdgcn_wmma_f32_16x16x32_bf16(
                false, fa0.v, false, fb0.v, (short)0, acc00, false, false);
            acc01 = __builtin_amdgcn_wmma_f32_16x16x32_bf16(
                false, fa0.v, false, fb1.v, (short)0, acc01, false, false);
            acc10 = __builtin_amdgcn_wmma_f32_16x16x32_bf16(
                false, fa1.v, false, fb0.v, (short)0, acc10, false, false);
            acc11 = __builtin_amdgcn_wmma_f32_16x16x32_bf16(
                false, fa1.v, false, fb1.v, (short)0, acc11, false, false);
        }
    }

    // --- epilogue: store fp32 conv output; accumulate GN sum / sumsq ---
    const int c0 = nt0 * 16 + nl16;
    const int c1 = c0 + 16;
    const size_t row0 = ((size_t)(b * COUT + c0) * Hs + yRow) * Ws + xBase;
    const size_t row1 = ((size_t)(b * COUT + c1) * Hs + yRow) * Ws + xBase;
    float s0 = 0.0f, q0 = 0.0f, s1 = 0.0f, q1 = 0.0f;
#pragma unroll
    for (int r = 0; r < 8; ++r) {
        const int mrow = r + hi * 8;
        const float v00 = acc00[r], v10 = acc10[r];      // group nt0
        const float v01 = acc01[r], v11 = acc11[r];      // group nt1
        out[row0 + mrow]      = v00;
        out[row0 + 16 + mrow] = v10;
        out[row1 + mrow]      = v01;
        out[row1 + 16 + mrow] = v11;
        s0 += v00 + v10;  q0 += v00 * v00 + v10 * v10;
        s1 += v01 + v11;  q1 += v01 * v01 + v11 * v11;
    }
#pragma unroll
    for (int d = 16; d > 0; d >>= 1) {
        s0 += __shfl_down(s0, d, 32);
        q0 += __shfl_down(q0, d, 32);
        s1 += __shfl_down(s1, d, 32);
        q1 += __shfl_down(q1, d, 32);
    }
    if (lane == 0) {
        atomicAdd(&ws[(b * GROUPS + nt0) * 2 + 0], s0);  // group == cout tile index
        atomicAdd(&ws[(b * GROUPS + nt0) * 2 + 1], q0);
        atomicAdd(&ws[(b * GROUPS + nt1) * 2 + 0], s1);
        atomicAdd(&ws[(b * GROUPS + nt1) * 2 + 1], q1);
    }
}

// ---------------------------------------------------------------------------
// Kernel 3: GroupNorm finalize, float4 vectorized in-place.
// ---------------------------------------------------------------------------
__global__ void __launch_bounds__(256)
gn_finalize_kernel(float* __restrict__ out,
                   const float* __restrict__ ws,
                   const float* __restrict__ gamma,
                   const float* __restrict__ beta) {
    const size_t i4 = (size_t)blockIdx.x * 256 + threadIdx.x;
    const size_t e  = i4 * 4;
    const int c = (int)((e / HW) % COUT);
    const int b = (int)(e / ((size_t)HW * COUT));
    const int g = c >> 4;
    const float s  = ws[(b * GROUPS + g) * 2 + 0];
    const float s2 = ws[(b * GROUPS + g) * 2 + 1];
    const float invN = 1.0f / (16.0f * (float)HW);
    const float mu  = s * invN;
    const float var = s2 * invN - mu * mu;
    const float rs  = rsqrtf(var + EPS);
    const float ga = gamma[c] * rs;
    const float be = beta[c];
    float4 v = *(float4*)&out[e];
    v.x = (v.x - mu) * ga + be;
    v.y = (v.y - mu) * ga + be;
    v.z = (v.z - mu) * ga + be;
    v.w = (v.w - mu) * ga + be;
    *(float4*)&out[e] = v;
}

// ---------------------------------------------------------------------------
extern "C" void kernel_launch(void* const* d_in, const int* in_sizes, int n_in,
                              void* d_out, int out_size, void* d_ws, size_t ws_size,
                              hipStream_t stream) {
    const float* x      = (const float*)d_in[0];
    const float* offset = (const float*)d_in[1];
    const float* mask   = (const float*)d_in[2];
    const float* weight = (const float*)d_in[3];
    const float* gamma  = (const float*)d_in[4];
    const float* beta   = (const float*)d_in[5];
    float* out = (float*)d_out;
    float* ws  = (float*)d_ws;
    unsigned int* pw = (unsigned int*)(ws + WS_ACC_FLOATS);

    zero_ws_kernel<<<1, 128, 0, stream>>>(ws);
    pack_weight_kernel<<<PW_DWORDS / 256, 256, 0, stream>>>(weight, pw);

    dcn_wmma_kernel<<<Bz * Hs * (Ws / 32), 256, 0, stream>>>(
        x, offset, mask, pw, out, ws);

    const int total4 = Bz * COUT * HW / 4;        // 2 359 296 float4
    gn_finalize_kernel<<<total4 / 256, 256, 0, stream>>>(out, ws, gamma, beta);
}
